// TransELayer_20409684590827
// MI455X (gfx1250) — compile-verified, matrix operand order checked by reading, not run on hardware
//
#include <hip/hip_runtime.h>
#include <hip/hip_bf16.h>
#include <cmath>

typedef __attribute__((ext_vector_type(16))) __bf16 v16bf;
typedef __attribute__((ext_vector_type(8)))  float  v8f;

#define HDIM   128
#define NNODES 10000
#define BATCH  128
#define TILE_N 40          // 250 blocks * 40 nodes = 10000
#define EPS    1e-12f

// ---------------- Kernel A: q[b] = normalize(node[ent[b]]) + normalize(rel[rel[b]]) ----------------
__global__ __launch_bounds__(256) void qnorm_kernel(
    const float* __restrict__ node_emb, const float* __restrict__ rel_emb,
    const int* __restrict__ ent, const int* __restrict__ rel,
    float* __restrict__ q)
{
    const int lane = threadIdx.x & 31;
    const int wave = threadIdx.x >> 5;
    const int b = blockIdx.x * 8 + wave;          // grid = 16 blocks -> 128 rows
    const float4 ev = ((const float4*)(node_emb + (size_t)ent[b] * HDIM))[lane];
    const float4 wv = ((const float4*)(rel_emb  + (size_t)rel[b] * HDIM))[lane];
    float se = ev.x*ev.x + ev.y*ev.y + ev.z*ev.z + ev.w*ev.w;
    float sw = wv.x*wv.x + wv.y*wv.y + wv.z*wv.z + wv.w*wv.w;
    #pragma unroll
    for (int off = 16; off; off >>= 1) {
        se += __shfl_xor(se, off, 32);
        sw += __shfl_xor(sw, off, 32);
    }
    const float re = 1.0f / fmaxf(sqrtf(se), EPS);
    const float rw = 1.0f / fmaxf(sqrtf(sw), EPS);
    float4 qv;
    qv.x = ev.x*re + wv.x*rw;  qv.y = ev.y*re + wv.y*rw;
    qv.z = ev.z*re + wv.z*rw;  qv.w = ev.w*re + wv.w*rw;
    ((float4*)(q + (size_t)b * HDIM))[lane] = qv;
}

// ---------------- Kernel B: rinv[n] = 1 / max(||node[n]||, eps) ----------------
__global__ __launch_bounds__(256) void rinv_kernel(
    const float* __restrict__ node_emb, float* __restrict__ rinv)
{
    const int lane = threadIdx.x & 31;
    const int wave = threadIdx.x >> 5;
    const int n = blockIdx.x * 8 + wave;
    if (n >= NNODES) return;
    const float4 v = ((const float4*)(node_emb + (size_t)n * HDIM))[lane];
    float s = v.x*v.x + v.y*v.y + v.z*v.z + v.w*v.w;
    #pragma unroll
    for (int off = 16; off; off >>= 1) s += __shfl_xor(s, off, 32);
    if (lane == 0) rinv[n] = 1.0f / fmaxf(sqrtf(s), EPS);
}

// ---------------- Kernel C: dist[b][n] = sum_h |q[b,h] - a[n,h]*rinv[n]| via bf16 WMMA ----------------
__global__ __launch_bounds__(256) void transe_dist_kernel(
    const float* __restrict__ node_emb,  // [N][H]
    const float* __restrict__ qbuf,      // [B][H]
    const float* __restrict__ rinv,      // [N]
    float* __restrict__ out)             // [B][N] distances
{
    __shared__ __attribute__((aligned(16))) float sA[TILE_N][HDIM];
    __shared__ float sR[TILE_N];

    const int tid  = threadIdx.x;
    const int lane = tid & 31;
    const int wave = tid >> 5;             // 8 waves -> 8 b-tiles of 16
    const int n0   = blockIdx.x * TILE_N;

    // Stage 40 node rows (20 KB) + rinv into LDS
    {
        const float4* src = (const float4*)(node_emb + (size_t)n0 * HDIM);
        float4* dst = (float4*)(&sA[0][0]);
        #pragma unroll
        for (int i = tid; i < TILE_N * HDIM / 4; i += 256) dst[i] = src[i];
        if (tid < TILE_N) sR[tid] = rinv[n0 + tid];
    }
    __syncthreads();

    // Per-lane WMMA A layout: M = lane&15; lanes<16 own K in {0..7,16..23} per chunk,
    // lanes>=16 own K in {8..15,24..31}  (ISA 16-bit A 16x32 layout).
    const int b0   = wave * 16;
    const int m    = lane & 15;
    const int koff = (lane >> 4) << 3;     // 0 or 8

    // Preload this lane's 64 q values (invariant over all n)
    float qreg[64];
    {
        const float* qrow = qbuf + (size_t)(b0 + m) * HDIM;
        #pragma unroll
        for (int c = 0; c < 4; ++c)
            #pragma unroll
            for (int oct = 0; oct < 2; ++oct) {
                const int h = 32*c + 16*oct + koff;
                const float4 v0 = *(const float4*)(qrow + h);
                const float4 v1 = *(const float4*)(qrow + h + 4);
                const int base = (c*2 + oct) * 8;
                qreg[base+0] = v0.x; qreg[base+1] = v0.y;
                qreg[base+2] = v0.z; qreg[base+3] = v0.w;
                qreg[base+4] = v1.x; qreg[base+5] = v1.y;
                qreg[base+6] = v1.z; qreg[base+7] = v1.w;
            }
    }

    // B matrix = ones (any layout works: all entries identical)
    v16bf bones;
    #pragma unroll
    for (int i = 0; i < 16; ++i) bones[i] = (__bf16)1.0f;

    for (int nl = 0; nl < TILE_N; ++nl) {
        const float nr = -sR[nl];          // d = fma(a, -rinv, q) = q - a/||a||
        const float* arow = &sA[nl][0];
        v8f acc = {};
        #pragma unroll
        for (int c = 0; c < 4; ++c) {
            // Convert SIGNED diffs to bf16, then clear sign bits in the packed
            // integer domain: bf16(|d|) == |bf16(d)| under RTNE. This avoids the
            // f32 fabs -> bf16-pack combine that crashes gfx1250 isel.
            union { v16bf v; unsigned int u[8]; } am;
            #pragma unroll
            for (int oct = 0; oct < 2; ++oct) {
                const int h = 32*c + 16*oct + koff;
                const float4 a0 = *(const float4*)(arow + h);       // LDS broadcast
                const float4 a1 = *(const float4*)(arow + h + 4);
                const int base = (c*2 + oct) * 8;
                am.v[oct*8+0] = (__bf16)fmaf(a0.x, nr, qreg[base+0]);
                am.v[oct*8+1] = (__bf16)fmaf(a0.y, nr, qreg[base+1]);
                am.v[oct*8+2] = (__bf16)fmaf(a0.z, nr, qreg[base+2]);
                am.v[oct*8+3] = (__bf16)fmaf(a0.w, nr, qreg[base+3]);
                am.v[oct*8+4] = (__bf16)fmaf(a1.x, nr, qreg[base+4]);
                am.v[oct*8+5] = (__bf16)fmaf(a1.y, nr, qreg[base+5]);
                am.v[oct*8+6] = (__bf16)fmaf(a1.z, nr, qreg[base+6]);
                am.v[oct*8+7] = (__bf16)fmaf(a1.w, nr, qreg[base+7]);
            }
            #pragma unroll
            for (int i = 0; i < 8; ++i) am.u[i] &= 0x7FFF7FFFu;     // |.| on packed bf16
            // D = A(|diff| bf16, 16x32) x B(ones) + C : every column = sum_h |diff|
            acc = __builtin_amdgcn_wmma_f32_16x16x32_bf16(
                false, am.v, false, bones, (short)0, acc, false, false);
        }
        // All 16 columns identical; lanes 0 and 16 hold rows 0..7 / 8..15 in acc[0..7]
        if ((lane & 15) == 0) {
            const int rbase = b0 + ((lane >> 4) << 3);
            #pragma unroll
            for (int v = 0; v < 8; ++v)
                out[(size_t)(rbase + v) * NNODES + (n0 + nl)] = acc[v];
        }
    }
}

// ---------------- Kernel D: in-place row softmax over N=10000 ----------------
__global__ __launch_bounds__(256) void softmax_kernel(float* __restrict__ out)
{
    __shared__ float red[8];
    float* row = out + (size_t)blockIdx.x * NNODES;
    const int tid = threadIdx.x;

    float m = -3.402823466e38f;
    for (int i = tid; i < NNODES; i += 256) m = fmaxf(m, row[i]);
    #pragma unroll
    for (int off = 16; off; off >>= 1) m = fmaxf(m, __shfl_xor(m, off, 32));
    if ((tid & 31) == 0) red[tid >> 5] = m;
    __syncthreads();
    float mall = red[0];
    #pragma unroll
    for (int i = 1; i < 8; ++i) mall = fmaxf(mall, red[i]);
    __syncthreads();

    float s = 0.f;
    for (int i = tid; i < NNODES; i += 256) s += expf(row[i] - mall);
    #pragma unroll
    for (int off = 16; off; off >>= 1) s += __shfl_xor(s, off, 32);
    if ((tid & 31) == 0) red[tid >> 5] = s;
    __syncthreads();
    float stot = 0.f;
    #pragma unroll
    for (int i = 0; i < 8; ++i) stot += red[i];
    const float inv = 1.0f / stot;

    for (int i = tid; i < NNODES; i += 256) row[i] = expf(row[i] - mall) * inv;
}

extern "C" void kernel_launch(void* const* d_in, const int* in_sizes, int n_in,
                              void* d_out, int out_size, void* d_ws, size_t ws_size,
                              hipStream_t stream)
{
    const float* node_emb = (const float*)d_in[0];   // [10000,128] f32
    const float* rel_emb  = (const float*)d_in[1];   // [500,128]   f32
    const int*   ent      = (const int*)d_in[2];     // [128] i32
    const int*   rel      = (const int*)d_in[3];     // [128] i32
    float* out = (float*)d_out;                      // [128,10000] f32

    float* q    = (float*)d_ws;                      // 128*128 f32 = 64 KB
    float* rinv = q + BATCH * HDIM;                  // 10000 f32   = 40 KB

    qnorm_kernel<<<BATCH / 8, 256, 0, stream>>>(node_emb, rel_emb, ent, rel, q);
    rinv_kernel<<<(NNODES + 7) / 8, 256, 0, stream>>>(node_emb, rinv);
    transe_dist_kernel<<<NNODES / TILE_N, 256, 0, stream>>>(node_emb, q, rinv, out);
    softmax_kernel<<<BATCH, 256, 0, stream>>>(out);
}
